// Linear_10402410791860
// MI455X (gfx1250) — matrix operand order repaired and checked by reading, not compiled, not run on hardware
//
#include <hip/hip_runtime.h>
#include <stdint.h>

typedef __attribute__((ext_vector_type(8)))  float  v8f;
typedef __attribute__((ext_vector_type(16))) __bf16 v16bf;

#define NPATH 15
#define ZT    16
#define DIM1  1152
#define DIM2  9
#define A2TOT 179

// ---- path tables (l1,l2,lo) in reference order -------------------------------
__device__ constexpr int PL1[NPATH] = {0,0,0,1,1,1,1,1,1,2,2,2,2,2,2};
__device__ constexpr int PL2[NPATH] = {0,1,2,0,1,1,1,2,2,0,1,1,2,2,2};
__device__ constexpr int PLO[NPATH] = {0,1,2,1,0,1,2,1,2,2,1,2,0,1,2};
__device__ constexpr int PI [NPATH] = {1,1,1,3,3,3,3,3,3,5,5,5,5,5,5};   // 2*l1+1
__device__ constexpr int PJ [NPATH] = {1,3,5,1,3,3,3,5,5,1,3,3,5,5,5};   // 2*l2+1
__device__ constexpr int PK [NPATH] = {1,3,5,3,1,3,5,3,5,5,3,5,1,3,5};   // 2*lo+1
__device__ constexpr int O1C[3]     = {0,128,512};       // x1/out column offset per l
__device__ constexpr int O2C[3]     = {0,1,4};           // x2 column offset per l
__device__ constexpr int A2OFF[NPATH] = {0,1,4,9,18,21,30,45,54,69,94,109,134,139,154};
__device__ constexpr int ACCB[3]    = {0,1,4};           // accumulator base per lo

struct CPtrs { const float* c[NPATH]; };

static __device__ __forceinline__ uint32_t pack_bf16(float a, float b) {
    union { __bf16 h[2]; uint32_t u; } v;
    v.h[0] = (__bf16)a; v.h[1] = (__bf16)b;
    return v.u;
}

// CDNA5 async global->LDS copy (GV mode), tracked by ASYNCcnt (ISA ch.10 / 15.18.3)
static __device__ __forceinline__ void async_load_b128(uint32_t lds_off, const void* gptr) {
    asm volatile("global_load_async_to_lds_b128 %0, %1, off"
                 :: "v"(lds_off), "v"(gptr)
                 : "memory");
}
static __device__ __forceinline__ void wait_asynccnt0() {
    asm volatile("s_wait_asynccnt 0" ::: "memory");
}

// ---- pre-kernel: fp32 weights [15][128][128] -> bf16 in WMMA B-fragment layout
// B (K=32 x N=16), contiguous k order used here: lane l -> column n=l%16; lanes 0-15 hold K=0..15,
// lanes 16-31 hold K=16..31; dword v packs K = base+2v (lo), base+2v+1 (hi).
// Scratch layout: dword index = (((p*8 + wtile)*4 + s)*32 + lane)*8 + v
__global__ __launch_bounds__(256)
void convert_weights(const float* __restrict__ ws, uint32_t* __restrict__ wbf) {
    int t = blockIdx.x * 256 + threadIdx.x;
    if (t >= NPATH * 8 * 4 * 32 * 8) return;
    int v  =  t        & 7;
    int L  = (t >> 3)  & 31;
    int s  = (t >> 8)  & 3;
    int wv = (t >> 10) & 7;
    int p  =  t >> 13;
    int w  = (wv << 4) + (L & 15);
    int u0 = (s << 5) + ((L >> 4) << 4) + (v << 1);
    const float* base = ws + p * 16384;          // ws[p][u][0][w]
    wbf[t] = pack_bf16(base[u0 * 128 + w], base[(u0 + 1) * 128 + w]);
}

// ---- main kernel -------------------------------------------------------------
__global__ __launch_bounds__(256)
void tp_kernel(const float* __restrict__ x1, const float* __restrict__ x2,
               CPtrs cs, const uint32_t* __restrict__ wbf,
               float* __restrict__ out, int n) {
    extern __shared__ char smem_raw[];
    float*    x1f  = (float*)smem_raw;                 // ZT*1152 fp32
    float*    x2f  = x1f + ZT * DIM1;                  // ZT*9
    float*    a2f  = x2f + ZT * DIM2;                  // ZT*179
    uint32_t* tbuf = (uint32_t*)(a2f + ZT * A2TOT);    // 5*4*32*8 dwords (A frags)

    const int tid   = threadIdx.x;
    const int lane  = tid & 31;
    const int wv    = tid >> 5;                        // wave id = w-tile id
    const int zbase = blockIdx.x * ZT;

    // ---- stage x1 tile asynchronously into LDS (no VGPR round-trip);
    //      overlapped with the a2 computation below (a2 depends only on x2/C).
    {
        const float4* src = (const float4*)(x1 + (size_t)zbase * DIM1);
        float4*       dst = (float4*)x1f;
        #pragma unroll
        for (int it = 0; it < (ZT * DIM1 / 4) / 256; ++it) {
            const int idx = tid + it * 256;
            async_load_b128((uint32_t)(uintptr_t)(const void*)(dst + idx), src + idx);
        }
        if (tid < ZT * DIM2) x2f[tid] = x2[(size_t)zbase * DIM2 + tid];
    }
    __syncthreads();    // x2f visible to all waves (x1 still in flight)

    // ---- a2_p[z,i,k] = sum_j C_p[i,j,k] * x2[z,j] (per-sample small tensors) --
    for (int f = tid; f < ZT * A2TOT; f += 256) {
        int z = f / A2TOT, r = f - z * A2TOT;
        float sum = 0.f;
        #pragma unroll
        for (int p = 0; p < NPATH; ++p) {
            int loc = r - A2OFF[p];
            if (loc >= 0 && loc < PI[p] * PK[p]) {
                int i = loc / PK[p], k = loc - i * PK[p];
                const float* cp = cs.c[p];
                const float* xr = x2f + z * DIM2 + O2C[PL2[p]];
                #pragma unroll
                for (int j = 0; j < PJ[p]; ++j)
                    sum += cp[(i * PJ[p] + j) * PK[p] + k] * xr[j];
            }
        }
        a2f[f] = sum;
    }

    wait_asynccnt0();   // this wave's x1 tile stripes are in LDS

    // ---- accumulators: 9 f32 16x16 tiles per wave: lo0:k0 | lo1:k0..2 | lo2:k0..4
    v8f acc[9];
    #pragma unroll
    for (int q = 0; q < 9; ++q)
        acc[q] = v8f{0.f,0.f,0.f,0.f,0.f,0.f,0.f,0.f};

    // ---- per-path: build T in LDS (A-fragment layout), then WMMA GEMM --------
    #pragma unroll
    for (int p = 0; p < NPATH; ++p) {
        const int I = PI[p], K = PK[p], LO = PLO[p];
        const int o1 = O1C[PL1[p]], a2o = A2OFF[p];

        __syncthreads();   // p==0: x1/a2 ready block-wide; p>0: tbuf consumers done

        // Produce T[z,u,k] as bf16 pairs in this code's contiguous k order:
        // lane<16: dword d -> K = 2d ; lane>=16: +16 (matches B fragment k order).
        const int iters = K * 1024;                    // K*4 frags * 256 dwords
        for (int f = tid; f < iters; f += 256) {
            int d = f & 7;
            int L = (f >> 3) & 31;
            int s = (f >> 8) & 3;
            int k = f >> 10;
            int z = L & 15;
            int e0 = d << 1;
            int K0 = ((L >> 4) << 4) + e0;
            int u0 = (s << 5) + K0;
            const float* xr = x1f + z * DIM1 + o1 + u0 * I;   // x1[z, u0, :]
            const float* ar = a2f + z * A2TOT + a2o + k;      // a2[z, i*K + k]
            float t0 = 0.f, t1 = 0.f;
            #pragma unroll
            for (int i = 0; i < I; ++i) {
                float a = ar[i * K];
                t0 += xr[i]     * a;
                t1 += xr[I + i] * a;
            }
            tbuf[((k * 4 + s) * 32 + L) * 8 + d] = pack_bf16(t0, t1);
        }
        __syncthreads();

        // B fragments for this path & this wave's w-tile (bf16, pre-swizzled, L2-hot)
        const uint32_t* wsrc = wbf + (((p * 8 + wv) * 4) * 32 + lane) * 8;
        v16bf b0 = *(const v16bf*)(wsrc + 0 * 256);
        v16bf b1 = *(const v16bf*)(wsrc + 1 * 256);
        v16bf b2 = *(const v16bf*)(wsrc + 2 * 256);
        v16bf b3 = *(const v16bf*)(wsrc + 3 * 256);

        #pragma unroll
        for (int k = 0; k < K; ++k) {
            v8f c = acc[ACCB[LO] + k];
            const uint32_t* tb = tbuf + (k * 4 * 32 + lane) * 8;
            v16bf a0 = *(const v16bf*)(tb + 0 * 256);
            v16bf a1 = *(const v16bf*)(tb + 1 * 256);
            v16bf a2 = *(const v16bf*)(tb + 2 * 256);
            v16bf a3 = *(const v16bf*)(tb + 3 * 256);
            c = __builtin_amdgcn_wmma_f32_16x16x32_bf16(false, a0, false, b0, (short)0, c, false, false);
            c = __builtin_amdgcn_wmma_f32_16x16x32_bf16(false, a1, false, b1, (short)0, c, false, false);
            c = __builtin_amdgcn_wmma_f32_16x16x32_bf16(false, a2, false, b2, (short)0, c, false, false);
            c = __builtin_amdgcn_wmma_f32_16x16x32_bf16(false, a3, false, b3, (short)0, c, false, false);
            acc[ACCB[LO] + k] = c;
        }
    }

    // ---- epilogue: normalize + store. D layout: lanes0-15 M=r, lanes16-31 M=8+r
    {
        constexpr int LO_OF[9] = {0,1,1,1,2,2,2,2,2};
        constexpr int K_OF [9] = {0,0,1,2,0,1,2,3,4};
        const float NRM[3] = {0.05103103630798288f,   // 1/sqrt(128*3)
                              0.03608439182435161f,   // 1/sqrt(128*6)
                              0.03608439182435161f};
        const int nloc = lane & 15;
        const int zh   = zbase + ((lane >> 4) << 3);
        #pragma unroll
        for (int q = 0; q < 9; ++q) {
            const int lo = LO_OF[q], k = K_OF[q];
            const int col = O1C[lo] + (wv * 16 + nloc) * (2 * lo + 1) + k;
            const float nr = NRM[lo];
            #pragma unroll
            for (int r = 0; r < 8; ++r)
                out[(size_t)(zh + r) * DIM1 + col] = acc[q][r] * nr;
        }
    }
    (void)n;
}

// ---- host entry --------------------------------------------------------------
extern "C" void kernel_launch(void* const* d_in, const int* in_sizes, int n_in,
                              void* d_out, int out_size, void* d_ws, size_t ws_size,
                              hipStream_t stream) {
    const float* x1 = (const float*)d_in[0];
    const float* x2 = (const float*)d_in[1];
    const float* ws = (const float*)d_in[2];
    CPtrs cp;
    for (int i = 0; i < NPATH; ++i) cp.c[i] = (const float*)d_in[3 + i];
    float*    out = (float*)d_out;
    uint32_t* wbf = (uint32_t*)d_ws;     // needs 491,520 B of scratch

    const int n = in_sizes[0] / DIM1;    // 16384
    const size_t smem = (size_t)(ZT * DIM1 + ZT * DIM2 + ZT * A2TOT) * 4
                      + (size_t)(5 * 4 * 32 * 8) * 4;   // 106,240 B (< 320 KB WGP LDS)

    (void)hipFuncSetAttribute((const void*)tp_kernel,
                              hipFuncAttributeMaxDynamicSharedMemorySize, (int)smem);

    convert_weights<<<(NPATH * 8192 + 255) / 256, 256, 0, stream>>>(ws, wbf);
    tp_kernel<<<n / ZT, 256, smem, stream>>>(x1, x2, cp, wbf, out, n);

    (void)n_in; (void)out_size; (void)ws_size;
}
